// SparseAttention_17970143167352
// MI455X (gfx1250) — matrix operand order repaired and checked by reading, not compiled, not run on hardware
//
#include <hip/hip_runtime.h>
#include <hip/hip_bf16.h>

typedef __attribute__((ext_vector_type(16))) _Float16 v16h;
typedef __attribute__((ext_vector_type(8)))  _Float16 v8h;
typedef __attribute__((ext_vector_type(2)))  _Float16 v2h;
typedef __attribute__((ext_vector_type(8)))  float    v8f;

#define IMG_H 56
#define IMG_W 56
#define HW    3136           // 56*56
#define CH    256
#define NTILE 196            // 3136 / 16
#define TILES_PER_MAT 6272   // 2 batches * 16 m-tiles * 196 n-tiles

// ---------------------------------------------------------------------------
// Kernel 1: fused QKV projection.  Y[o,p] = sum_c W[o,c] * X[c,p] + bias[o]
// X: channel-major f32 [b][256][3136].  Output: PIXEL-major f16 [b][3136][256]
// so each lane's 8 C-layout rows (consecutive o) pack into one 16B store.
// One wave32 per 16x16 tile, 8x v_wmma_f32_16x16x32_f16 over K=256.
// ---------------------------------------------------------------------------
__global__ __launch_bounds__(256) void proj_qkv_wmma(
    const float* __restrict__ X,
    const float* __restrict__ W0, const float* __restrict__ Bi0, _Float16* __restrict__ Y0,
    const float* __restrict__ W1, const float* __restrict__ Bi1, _Float16* __restrict__ Y1,
    const float* __restrict__ W2, const float* __restrict__ Bi2, _Float16* __restrict__ Y2)
{
    const int wid  = blockIdx.x * 8 + (threadIdx.x >> 5);
    const int lane = threadIdx.x & 31;
    const int half = lane >> 4;
    const int lm   = lane & 15;

    const int mat = wid / TILES_PER_MAT;
    int r = wid % TILES_PER_MAT;
    const int b  = r / 3136;
    r %= 3136;
    const int m0 = (r / NTILE) * 16;
    const int p0 = (r % NTILE) * 16;

    const float* W  = (mat == 0) ? W0  : (mat == 1) ? W1  : W2;
    const float* Bi = (mat == 0) ? Bi0 : (mat == 1) ? Bi1 : Bi2;
    _Float16*    Y  = (mat == 0) ? Y0  : (mat == 1) ? Y1  : Y2;

    const float* Xb = X + (size_t)b * CH * HW;

    v8f acc = {};
    #pragma unroll
    for (int k0 = 0; k0 < CH; k0 += 32) {
        v16h a, bb;
        #pragma unroll
        for (int v = 0; v < 8; ++v) {
            // A (16x32, 16-bit): lane = row M=lm; K = k0+16*(v>>2)+8*half+2*(v&3)+{0,1}
            const int ka = k0 + 16 * (v >> 2) + 8 * half + 2 * (v & 3);
            const float2 w2 = *reinterpret_cast<const float2*>(W + (size_t)(m0 + lm) * CH + ka);
            a[2 * v]     = (_Float16)w2.x;
            a[2 * v + 1] = (_Float16)w2.y;
            // B (32x16, 16-bit): lane = col N=lm; K = k0+16*(v>>2)+8*half+2*(v&3)+{0,1}
            const int kb = k0 + 16 * (v >> 2) + 8 * half + 2 * (v & 3);
            bb[2 * v]     = (_Float16)Xb[(size_t)kb * HW + p0 + lm];
            bb[2 * v + 1] = (_Float16)Xb[(size_t)(kb + 1) * HW + p0 + lm];
        }
        acc = __builtin_amdgcn_wmma_f32_16x16x32_f16(false, a, false, bb,
                                                     (short)0, acc, false, false);
    }
    // C/D f32 16x16: VGPR rr -> M = m0 + 8*half + rr (consecutive!), lane -> p0+lm.
    // Pack 8 consecutive output channels into one 16B pixel-major store.
    v8h dpack;
    #pragma unroll
    for (int rr = 0; rr < 8; ++rr)
        dpack[rr] = (_Float16)(acc[rr] + Bi[m0 + 8 * half + rr]);
    const size_t pixbase = ((size_t)b * HW + p0 + lm) * CH;
    *reinterpret_cast<v8h*>(Y + pixbase + m0 + 8 * half) = dpack;
}

// ---------------------------------------------------------------------------
// Kernel 2: 7x7 sliding-window attention, one (pixel, head) per thread.
// K/V halo pixels are 64B contiguous in the pixel-major f16 layout, streamed
// straight into LDS with global_load_async_to_lds_b128 (ASYNCcnt tracked);
// zero-padding borders written with 16B ds stores.
// ---------------------------------------------------------------------------
#define TY     2
#define HALOH  8            // TY + 6
#define HALOW  62           // 56 + 6
#define DPP    32           // head dim = LDS pitch (64B per pixel)

__global__ __launch_bounds__(128) void win_attn(
    const _Float16* __restrict__ QT, const _Float16* __restrict__ KT,
    const _Float16* __restrict__ VT, _Float16* __restrict__ OT)
{
    __shared__ __align__(16) _Float16 ksm[HALOH * HALOW * DPP];  // 31744 B
    __shared__ __align__(16) _Float16 vsm[HALOH * HALOW * DPP];  // 31744 B

    const int tid = threadIdx.x;
    const int y0  = blockIdx.x * TY;
    const int h   = blockIdx.y;
    const int b   = blockIdx.z;

    // halo fill: 8*62 pixels * 4 quarters (16B each), interior via async-to-LDS
    const int NT = HALOH * HALOW * 4;
    for (int i = tid; i < NT; i += 128) {
        const int quarter = i & 3;
        const int pix = i >> 2;
        const int xx = pix % HALOW;
        const int ry = pix / HALOW;
        const int gy = y0 + ry - 3;
        const int gx = xx - 3;
        const int loff = pix * DPP + quarter * 8;            // halfs
        if (gy >= 0 && gy < IMG_H && gx >= 0 && gx < IMG_W) {
            const size_t goff =
                ((size_t)b * HW + (size_t)gy * IMG_W + gx) * CH + h * DPP + quarter * 8;
            const unsigned kl = (unsigned)(uintptr_t)&ksm[loff];
            const unsigned vl = (unsigned)(uintptr_t)&vsm[loff];
            asm volatile("global_load_async_to_lds_b128 %0, %1, off"
                         :: "v"(kl), "v"(KT + goff) : "memory");
            asm volatile("global_load_async_to_lds_b128 %0, %1, off"
                         :: "v"(vl), "v"(VT + goff) : "memory");
        } else {
            const v8h z = {};
            *reinterpret_cast<v8h*>(&ksm[loff]) = z;
            *reinterpret_cast<v8h*>(&vsm[loff]) = z;
        }
    }
    asm volatile("s_wait_asynccnt 0" ::: "memory");
    __syncthreads();

    if (tid < IMG_W * TY) {
        const int x  = tid % IMG_W;
        const int yy = tid / IMG_W;
        const int y  = y0 + yy;
        const size_t pixbase = ((size_t)b * HW + (size_t)y * IMG_W + x) * CH + h * DPP;

        float q[DPP];
        const v8h* qp = reinterpret_cast<const v8h*>(QT + pixbase);
        #pragma unroll
        for (int vq = 0; vq < 4; ++vq) {
            const v8h t = qp[vq];
            #pragma unroll
            for (int d = 0; d < 8; ++d) q[vq * 8 + d] = (float)t[d];
        }

        const float scale = 0.17677669529663687f;   // 32^-0.5
        float s[49];
        #pragma unroll
        for (int i = 0; i < 7; ++i) {
            #pragma unroll
            for (int j = 0; j < 7; ++j) {
                const v8h* kp = reinterpret_cast<const v8h*>(
                    &ksm[((yy + i) * HALOW + (x + j)) * DPP]);
                float a = 0.0f;
                #pragma unroll
                for (int v = 0; v < 4; ++v) {
                    const v8h kk = kp[v];
                    #pragma unroll
                    for (int d = 0; d < 8; ++d)
                        a += q[v * 8 + d] * (float)kk[d];
                }
                s[i * 7 + j] = a * scale;
            }
        }
        float m = s[0];
        #pragma unroll
        for (int w = 1; w < 49; ++w) m = fmaxf(m, s[w]);
        float sum = 0.0f;
        #pragma unroll
        for (int w = 0; w < 49; ++w) { s[w] = __expf(s[w] - m); sum += s[w]; }
        const float inv = 1.0f / sum;

        float o[DPP];
        #pragma unroll
        for (int d = 0; d < DPP; ++d) o[d] = 0.0f;
        #pragma unroll
        for (int i = 0; i < 7; ++i) {
            #pragma unroll
            for (int j = 0; j < 7; ++j) {
                const v8h* vp = reinterpret_cast<const v8h*>(
                    &vsm[((yy + i) * HALOW + (x + j)) * DPP]);
                const float a = s[i * 7 + j];
                #pragma unroll
                for (int v = 0; v < 4; ++v) {
                    const v8h vk = vp[v];
                    #pragma unroll
                    for (int d = 0; d < 8; ++d)
                        o[v * 8 + d] += a * (float)vk[d];
                }
            }
        }
        v8h* op = reinterpret_cast<v8h*>(OT + pixbase);
        #pragma unroll
        for (int vq = 0; vq < 4; ++vq) {
            v8h t;
            #pragma unroll
            for (int d = 0; d < 8; ++d) t[d] = (_Float16)(o[vq * 8 + d] * inv);
            op[vq] = t;
        }
    }
}

// ---------------------------------------------------------------------------
// Kernel 3: output projection.  X: pixel-major f16 [b][3136][256] (no cvts on
// the B operand -- contiguous f16 pairs, mergeable to b128), output: channel-
// major f32 [b][256][3136] = d_out, with bias.
// ---------------------------------------------------------------------------
__global__ __launch_bounds__(256) void proj_out_wmma(
    const _Float16* __restrict__ XT,
    const float* __restrict__ W, const float* __restrict__ Bi,
    float* __restrict__ Y)
{
    const int wid  = blockIdx.x * 8 + (threadIdx.x >> 5);
    const int lane = threadIdx.x & 31;
    const int half = lane >> 4;
    const int lm   = lane & 15;

    int r = wid;                       // 0 .. 6271
    const int b  = r / 3136;
    r %= 3136;
    const int m0 = (r / NTILE) * 16;
    const int p0 = (r % NTILE) * 16;

    const size_t xb = ((size_t)b * HW + p0 + lm) * CH;
    float* Yb = Y + (size_t)b * CH * HW;

    v8f acc = {};
    #pragma unroll
    for (int k0 = 0; k0 < CH; k0 += 32) {
        v16h a, bb;
        #pragma unroll
        for (int v = 0; v < 8; ++v) {
            const int ka = k0 + 16 * (v >> 2) + 8 * half + 2 * (v & 3);
            const float2 w2 = *reinterpret_cast<const float2*>(W + (size_t)(m0 + lm) * CH + ka);
            a[2 * v]     = (_Float16)w2.x;
            a[2 * v + 1] = (_Float16)w2.y;
            const int kb = k0 + 16 * (v >> 2) + 8 * half + 2 * (v & 3);
            const v2h hv = *reinterpret_cast<const v2h*>(XT + xb + kb);
            bb[2 * v]     = hv[0];
            bb[2 * v + 1] = hv[1];
        }
        acc = __builtin_amdgcn_wmma_f32_16x16x32_f16(false, a, false, bb,
                                                     (short)0, acc, false, false);
    }
    #pragma unroll
    for (int rr = 0; rr < 8; ++rr) {
        const int o = m0 + 8 * half + rr;
        Yb[(size_t)o * HW + p0 + lm] = acc[rr] + Bi[o];
    }
}

// ---------------------------------------------------------------------------
extern "C" void kernel_launch(void* const* d_in, const int* in_sizes, int n_in,
                              void* d_out, int out_size, void* d_ws, size_t ws_size,
                              hipStream_t stream) {
    const float* x  = (const float*)d_in[0];
    const float* wq = (const float*)d_in[1];
    const float* bq = (const float*)d_in[2];
    const float* wk = (const float*)d_in[3];
    const float* bk = (const float*)d_in[4];
    const float* wv = (const float*)d_in[5];
    const float* bv = (const float*)d_in[6];
    const float* wo = (const float*)d_in[7];
    const float* bo = (const float*)d_in[8];
    float* out = (float*)d_out;

    const size_t per = (size_t)2 * HW * CH;       // elems per f16 tensor (3.2 MB)
    _Float16* qT  = (_Float16*)d_ws;
    _Float16* kT  = qT + per;
    _Float16* vT  = kT + per;
    _Float16* aoT = vT + per;

    // 1) fused Q/K/V projections -> pixel-major f16
    proj_qkv_wmma<<<(3 * TILES_PER_MAT) / 8, 256, 0, stream>>>(
        x, wq, bq, qT, wk, bk, kT, wv, bv, vT);

    // 2) windowed attention (async-to-LDS K/V staging)
    win_attn<<<dim3(IMG_H / TY, 8, 2), 128, 0, stream>>>(qT, kT, vT, aoT);

    // 3) output projection -> channel-major f32 d_out
    proj_out_wmma<<<TILES_PER_MAT / 8, 256, 0, stream>>>(aoT, wo, bo, out);
}
